// MonoSFNet_22608707846219
// MI455X (gfx1250) — matrix operand order repaired and checked by reading, not compiled, not run on hardware
//
#include <hip/hip_runtime.h>
#include <hip/hip_bf16.h>
#include <cstdint>
#include <cstddef>

typedef _Float16 half_t;
typedef __attribute__((ext_vector_type(16))) _Float16 v16h;
typedef __attribute__((ext_vector_type(8)))  float    v8f;

#define BN_S 0.99999500003749968f   /* 1/sqrt(1+1e-5), eval-mode BN */
#define ACT_NONE  0
#define ACT_LRELU 1
#define ACT_RELU  2

__device__ __forceinline__ int imin(int a, int b) { return a < b ? a : b; }
__device__ __forceinline__ int imax(int a, int b) { return a > b ? a : b; }

// ---------------------------------------------------------------------------
// Weight pack: OIHW f32 -> WMMA A-fragment (16x32 f16) layout per (coTile, kChunk)
// Lane l holds row m = l&15;  K(l,h) = kc*32 + 16*(l>>4) + h
// ---------------------------------------------------------------------------
__global__ __launch_bounds__(32) void k_pack_w(
    const float* __restrict__ w, half_t* __restrict__ wpack,
    int Cout, int Cin, int ks, int kChunks)
{
    const int lane = threadIdx.x;
    const int kc = blockIdx.x % kChunks;
    const int t  = blockIdx.x / kChunks;
    const int Kdim = Cin * ks * ks;
    const int m  = t * 16 + (lane & 15);
    const int kb = (lane >> 4) * 8;
    half_t* dst = wpack + ((size_t)(t * kChunks + kc) * 32 + lane) * 16;
#pragma unroll
    for (int h = 0; h < 16; ++h) {
        int k = kc * 32 + 2 * kb + h;
        float v = 0.f;
        if (m < Cout && k < Kdim) v = w[(size_t)m * Kdim + k];
        dst[h] = (half_t)v;
    }
}

// ---------------------------------------------------------------------------
// Implicit-GEMM conv via V_WMMA_F32_16X16X32_F16, templated on kernel size so
// the im2col K-decode strength-reduces to shifts/mul (no runtime int division).
// One wave computes a 16(Cout) x 16(pixel) tile; K = Cin*KS*KS in chunks of 32.
// B-fragment (32x16 KxN): lane l holds col n = l&15, K = 16*(l>>4) + h.
// Epilogue: fused eval-BN, bias, activation; channel-offset output (free concat).
// ---------------------------------------------------------------------------
template <int KS>
__global__ __launch_bounds__(32) void k_conv_wmma(
    const float* __restrict__ x, int xCtot, int xOff,
    const half_t* __restrict__ wpack,
    const float* __restrict__ bias,
    const float* __restrict__ bnG, const float* __restrict__ bnB,
    float* __restrict__ y, int yCtot, int yOff,
    int Cin, int H, int W,
    int Cout, int Hout, int Wout,
    int stride, int dil, int pad,
    int kChunks, int act)
{
    const int lane   = threadIdx.x;
    const int coTile = blockIdx.x;
    const int pixT   = blockIdx.y;
    const int b      = blockIdx.z;
    const int npix   = Hout * Wout;
    constexpr int kk = KS * KS;
    const int Kdim   = Cin * kk;

    const int nIdx = pixT * 16 + (lane & 15);
    const int n    = imin(nIdx, npix - 1);
    const int oh   = n / Wout, ow = n - (n / Wout) * Wout;
    const int ihB  = oh * stride - pad;          // KS==1: pad==0, stride==1 here
    const int iwB  = ow * stride - pad;
    const int kbB  = (lane >> 4) * 16;
    const size_t HW = (size_t)H * W;

    const float* xb = x + ((size_t)b * xCtot + xOff) * HW;

    v8f acc = {};
    const half_t* wp = wpack + (size_t)coTile * kChunks * 32 * 16 + (size_t)lane * 16;

#pragma unroll 2
    for (int kc = 0; kc < kChunks; ++kc) {
        v16h afrag = *(const v16h*)wp;
        wp += 32 * 16;
        __builtin_prefetch(wp, 0, 1);   // global_prefetch_b8 on next weight chunk

        v16h bfrag;
        const int k0 = kc * 32 + kbB;
#pragma unroll
        for (int h = 0; h < 16; ++h) {
            int k = k0 + h;
            float v = 0.f;
            if (KS == 1) {
                // k == ci; stride==1, pad==0 for every 1x1 conv in this net
                if (k < Kdim) v = xb[(size_t)k * HW + n];
            } else {
                if (k < Kdim) {
                    int ci = k / kk;             // compile-time divisor -> mul/shift
                    int r  = k - ci * kk;
                    int kh = r / KS, kw = r - kh * KS;
                    int iy = ihB + kh * dil;
                    int ix = iwB + kw * dil;
                    if (iy >= 0 && iy < H && ix >= 0 && ix < W)
                        v = xb[(size_t)ci * HW + (size_t)iy * W + ix];
                }
            }
            bfrag[h] = (half_t)v;
        }
        acc = __builtin_amdgcn_wmma_f32_16x16x32_f16(
            /*neg_a=*/false, afrag, /*neg_b=*/false, bfrag,
            /*c_mod=*/(short)0, acc, /*reuse_a=*/false, /*reuse_b=*/false);
    }

    if (nIdx < npix) {
        const int mrow = (lane >> 4) * 8;
#pragma unroll
        for (int r = 0; r < 8; ++r) {
            int m = coTile * 16 + mrow + r;
            if (m < Cout) {
                float v = acc[r];
                if (bnG)  v = v * (bnG[m] * BN_S) + bnB[m];
                if (bias) v += bias[m];
                if (act == ACT_LRELU) v = v >= 0.f ? v : 0.1f * v;
                else if (act == ACT_RELU) v = fmaxf(v, 0.f);
                y[((size_t)b * yCtot + yOff + m) * (size_t)npix + nIdx] = v;
            }
        }
    }
}

template __global__ void k_conv_wmma<1>(
    const float*, int, int, const half_t*, const float*, const float*, const float*,
    float*, int, int, int, int, int, int, int, int, int, int, int, int, int);
template __global__ void k_conv_wmma<3>(
    const float*, int, int, const half_t*, const float*, const float*, const float*,
    float*, int, int, int, int, int, int, int, int, int, int, int, int, int);

// --------------------------- depthwise 3x3 + BN + act ----------------------
__global__ void k_dwconv3x3(
    const float* __restrict__ x, int xCtot, int xOff,
    const float* __restrict__ w,
    const float* __restrict__ bnG, const float* __restrict__ bnB,
    float* __restrict__ y, int yCtot, int yOff,
    int B, int C, int H, int W, int act)
{
    int idx = blockIdx.x * blockDim.x + threadIdx.x;
    int total = B * C * H * W;
    if (idx >= total) return;
    int ww = idx % W; int t = idx / W;
    int hh = t % H;   t /= H;
    int c  = t % C;   int b = t / C;
    const float* xc = x + ((size_t)b * xCtot + xOff + c) * (size_t)H * W;
    const float* wc = w + c * 9;
    float acc = 0.f;
#pragma unroll
    for (int kh = 0; kh < 3; ++kh) {
        int iy = hh + kh - 1;
        if (iy < 0 || iy >= H) continue;
#pragma unroll
        for (int kw = 0; kw < 3; ++kw) {
            int ix = ww + kw - 1;
            if (ix < 0 || ix >= W) continue;
            acc += wc[kh * 3 + kw] * xc[(size_t)iy * W + ix];
        }
    }
    if (bnG) acc = acc * (bnG[c] * BN_S) + bnB[c];
    if (act == ACT_LRELU) acc = acc >= 0.f ? acc : 0.1f * acc;
    else if (act == ACT_RELU) acc = fmaxf(acc, 0.f);
    y[((size_t)b * yCtot + yOff + c) * (size_t)H * W + (size_t)hh * W + ww] = acc;
}

// ------------------- 81-channel cost volume (+LeakyReLU) -------------------
__global__ void k_corr81(
    const float* __restrict__ f1, const float* __restrict__ f2,
    float* __restrict__ y, int yCtot, int yOff,
    int B, int C, int H, int W)
{
    int idx = blockIdx.x * blockDim.x + threadIdx.x;
    int total = B * H * W;
    if (idx >= total) return;
    int w = idx % W; int t = idx / W;
    int h = t % H;   int b = t / H;
    const float* f1b = f1 + (size_t)b * C * H * W;
    const float* f2b = f2 + (size_t)b * C * H * W;
    const float invC = 1.0f / (float)C;
    int d = 0;
    for (int dy = -4; dy <= 4; ++dy) {
        for (int dx = -4; dx <= 4; ++dx, ++d) {
            float s = 0.f;
            int yy = h + dy, xx = w + dx;
            if (yy >= 0 && yy < H && xx >= 0 && xx < W) {
                for (int c = 0; c < C; ++c)
                    s += f1b[((size_t)c * H + h) * W + w] * f2b[((size_t)c * H + yy) * W + xx];
            }
            s *= invC;
            s = s >= 0.f ? s : 0.1f * s;
            y[((size_t)b * yCtot + yOff + d) * (size_t)H * W + (size_t)h * W + w] = s;
        }
    }
}

// ---------------- bilinear warp, border clamp, align-corners ---------------
__global__ void k_warp(
    const float* __restrict__ x, const float* __restrict__ flow,
    float* __restrict__ y, int B, int C, int H, int W)
{
    int idx = blockIdx.x * blockDim.x + threadIdx.x;
    int total = B * H * W;
    if (idx >= total) return;
    int w = idx % W; int t = idx / W;
    int h = t % H;   int b = t / H;
    size_t hw = (size_t)H * W;
    float fx = flow[((size_t)b * 2 + 0) * hw + (size_t)h * W + w];
    float fy = flow[((size_t)b * 2 + 1) * hw + (size_t)h * W + w];
    float px = fminf(fmaxf((float)w + fx, 0.f), (float)(W - 1));
    float py = fminf(fmaxf((float)h + fy, 0.f), (float)(H - 1));
    int x0 = (int)floorf(px), y0 = (int)floorf(py);
    float wx = px - (float)x0, wy = py - (float)y0;
    int x1 = imin(x0 + 1, W - 1), y1 = imin(y0 + 1, H - 1);
    float w00 = (1.f - wx) * (1.f - wy), w01 = wx * (1.f - wy);
    float w10 = (1.f - wx) * wy,         w11 = wx * wy;
    const float* xb = x + (size_t)b * C * hw;
    float* yb = y + (size_t)b * C * hw;
    for (int c = 0; c < C; ++c) {
        const float* xc = xb + (size_t)c * hw;
        yb[(size_t)c * hw + (size_t)h * W + w] =
            xc[(size_t)y0 * W + x0] * w00 + xc[(size_t)y0 * W + x1] * w01 +
            xc[(size_t)y1 * W + x0] * w10 + xc[(size_t)y1 * W + x1] * w11;
    }
}

// ----------------- bilinear resize, align_corners=True, *mul ---------------
__global__ void k_resize_ac(
    const float* __restrict__ x, float* __restrict__ y,
    int B, int C, int Hi, int Wi, int Ho, int Wo, float mul)
{
    int idx = blockIdx.x * blockDim.x + threadIdx.x;
    int total = B * C * Ho * Wo;
    if (idx >= total) return;
    int wo = idx % Wo; int t = idx / Wo;
    int ho = t % Ho;   t /= Ho;
    int c  = t % C;    int b = t / C;
    float rh = (Ho > 1) ? (float)(Hi - 1) / (float)(Ho - 1) : 0.f;
    float rw = (Wo > 1) ? (float)(Wi - 1) / (float)(Wo - 1) : 0.f;
    float ph = (float)ho * rh, pw = (float)wo * rw;
    int h0 = (int)ph, w0 = (int)pw;
    int h1 = imin(h0 + 1, Hi - 1), w1 = imin(w0 + 1, Wi - 1);
    float ay = ph - (float)h0, ax = pw - (float)w0;
    const float* xc = x + ((size_t)b * C + c) * (size_t)Hi * Wi;
    float v = (xc[(size_t)h0 * Wi + w0] * (1.f - ax) + xc[(size_t)h0 * Wi + w1] * ax) * (1.f - ay)
            + (xc[(size_t)h1 * Wi + w0] * (1.f - ax) + xc[(size_t)h1 * Wi + w1] * ax) * ay;
    y[idx] = v * mul;
}

// ----------------------------- nearest 2x upsample -------------------------
__global__ void k_nearest2x(
    const float* __restrict__ x, int xCtot, int xOff,
    float* __restrict__ y, int yCtot, int yOff,
    int B, int C, int Hi, int Wi)
{
    int Ho = Hi * 2, Wo = Wi * 2;
    int idx = blockIdx.x * blockDim.x + threadIdx.x;
    int total = B * C * Ho * Wo;
    if (idx >= total) return;
    int wo = idx % Wo; int t = idx / Wo;
    int ho = t % Ho;   t /= Ho;
    int c  = t % C;    int b = t / C;
    float v = x[((size_t)b * xCtot + xOff + c) * (size_t)Hi * Wi + (size_t)(ho >> 1) * Wi + (wo >> 1)];
    y[((size_t)b * yCtot + yOff + c) * (size_t)Ho * Wo + (size_t)ho * Wo + wo] = v;
}

// ------------------------------ channel-slice copy -------------------------
__global__ void k_copy_ch(
    const float* __restrict__ x, int xCtot, int xOff,
    float* __restrict__ y, int yCtot, int yOff,
    int B, int C, int H, int W)
{
    int idx = blockIdx.x * blockDim.x + threadIdx.x;
    int total = B * C * H * W;
    if (idx >= total) return;
    int p = idx % (H * W); int t = idx / (H * W);
    int c = t % C; int b = t / C;
    y[((size_t)b * yCtot + yOff + c) * (size_t)H * W + p] =
        x[((size_t)b * xCtot + xOff + c) * (size_t)H * W + p];
}

__global__ void k_add(float* __restrict__ dst, const float* __restrict__ src, int n)
{
    int i = blockIdx.x * blockDim.x + threadIdx.x;
    if (i < n) dst[i] += src[i];
}

__global__ void k_fill(float* __restrict__ p, float v, int n)
{
    int i = blockIdx.x * blockDim.x + threadIdx.x;
    if (i < n) p[i] = v;
}

// ===========================================================================
// Host orchestration
// ===========================================================================
namespace {

struct Ws {
    char* base; size_t size; size_t off;
    void* take(size_t bytes) {
        bytes = (bytes + 255) & ~(size_t)255;
        if (bytes > size) bytes = size;
        if (off + bytes > size) off = 0;   // wrap guard (ws_size opaque)
        void* p = base + off; off += bytes; return p;
    }
    float*  f(size_t n) { return (float*)take(n * sizeof(float)); }
    half_t* h(size_t n) { return (half_t*)take(n * sizeof(half_t)); }
};

struct ConvW { const float* w; const float* bias; int Cin, Cout, ks, kChunks; half_t* pack; };

static ConvW mkConv(Ws& ws, hipStream_t s, const float* w, const float* bias,
                    int Cin, int Cout, int ks)
{
    ConvW c; c.w = w; c.bias = bias; c.Cin = Cin; c.Cout = Cout; c.ks = ks;
    c.kChunks = (Cin * ks * ks + 31) / 32;
    int coT = (Cout + 15) / 16;
    c.pack = ws.h((size_t)coT * c.kChunks * 32 * 16);
    k_pack_w<<<dim3(coT * c.kChunks), dim3(32), 0, s>>>(w, c.pack, Cout, Cin, ks, c.kChunks);
    return c;
}

static inline int outDim(int H, int ks, int stride, int dil) {
    int pad = ((ks - 1) * dil) / 2;
    return (H + 2 * pad - (dil * (ks - 1) + 1)) / stride + 1;
}

static void runConv(hipStream_t s, const ConvW& c,
                    const float* x, int xCtot, int xOff, int H, int W,
                    float* y, int yCtot, int yOff, int B,
                    int stride, int dil,
                    const float* bnG, const float* bnB, int act)
{
    int pad = ((c.ks - 1) * dil) / 2;
    int Ho = outDim(H, c.ks, stride, dil);
    int Wo = outDim(W, c.ks, stride, dil);
    int npix = Ho * Wo;
    dim3 g((c.Cout + 15) / 16, (npix + 15) / 16, B);
    if (c.ks == 1)
        k_conv_wmma<1><<<g, dim3(32), 0, s>>>(
            x, xCtot, xOff, c.pack, c.bias, bnG, bnB,
            y, yCtot, yOff, c.Cin, H, W, c.Cout, Ho, Wo,
            stride, dil, pad, c.kChunks, act);
    else
        k_conv_wmma<3><<<g, dim3(32), 0, s>>>(
            x, xCtot, xOff, c.pack, c.bias, bnG, bnB,
            y, yCtot, yOff, c.Cin, H, W, c.Cout, Ho, Wo,
            stride, dil, pad, c.kChunks, act);
}

static inline dim3 g1(size_t n) { return dim3((unsigned)((n + 255) / 256)); }

} // namespace

extern "C" void kernel_launch(void* const* d_in, const int* in_sizes, int n_in,
                              void* d_out, int out_size, void* d_ws, size_t ws_size,
                              hipStream_t stream)
{
    (void)in_sizes; (void)n_in; (void)out_size;
    Ws ws{(char*)d_ws, ws_size, 0};
    const int B = 2, H0 = 192, W0 = 640;

    // -------- unpack inputs: im1, im2, then params in jax tree_leaves order
    // (dict keys sorted: c11, cls, ctx, fe, fpe, seg; None bias leaves dropped)
    int cur = 0;
    auto F = [&]() -> const float* { return (const float*)d_in[cur++]; };

    const float* im1 = F();
    const float* im2 = F();

    const float *c11_b[5], *c11_w[5];
    for (int i = 0; i < 5; ++i) { c11_b[i] = F(); c11_w[i] = F(); }

    struct Dsc { const float *bn1_b,*bn1_g,*bn2_b,*bn2_g,*dw_w,*pw_w; } d1, d2;
    auto rdDsc = [&](Dsc& d) { d.bn1_b=F(); d.bn1_g=F(); d.bn2_b=F(); d.bn2_g=F(); d.dw_w=F(); d.pw_w=F(); };
    rdDsc(d1); rdDsc(d2);
    const float* cls_out_b = F(); const float* cls_out_w = F();

    const float *ctx_b[7], *ctx_w[7];
    for (int i = 0; i < 7; ++i) { ctx_b[i] = F(); ctx_w[i] = F(); }

    const float *fe_b[6], *fe_w[6];
    for (int i = 0; i < 6; ++i) { fe_b[i] = F(); fe_w[i] = F(); }

    const float *fpe1_b[6], *fpe1_w[6], *fpe2_b[6], *fpe2_w[6];
    for (int i = 0; i < 6; ++i) { fpe1_b[i]=F(); fpe1_w[i]=F(); fpe2_b[i]=F(); fpe2_w[i]=F(); }

    struct SegL { const float *bnd_b,*bnd_g,*bne_b,*bne_g,*bnp_b,*bnp_g,*dw_w,*exp_w,*proj_w; } seg[6];
    for (int i = 0; i < 6; ++i) {
        SegL& sl = seg[i];
        sl.bnd_b=F(); sl.bnd_g=F(); sl.bne_b=F(); sl.bne_g=F(); sl.bnp_b=F(); sl.bnp_g=F();
        sl.dw_w=F(); sl.exp_w=F(); sl.proj_w=F();
    }

    // -------- pack all dense-conv weights to WMMA A-fragment layout
    static const int chs[7] = {3, 16, 32, 64, 96, 128, 192};
    ConvW cFpe1[6], cFpe2[6];
    for (int i = 0; i < 6; ++i) {
        cFpe1[i] = mkConv(ws, stream, fpe1_w[i], fpe1_b[i], chs[i],   chs[i+1], 3);
        cFpe2[i] = mkConv(ws, stream, fpe2_w[i], fpe2_b[i], chs[i+1], chs[i+1], 3);
    }
    static const int c11ci[5] = {192, 128, 96, 64, 32};
    ConvW cC11[5];
    for (int i = 0; i < 5; ++i) cC11[i] = mkConv(ws, stream, c11_w[i], c11_b[i], c11ci[i], 32, 1);

    static const int feCi[6] = {115, 243, 371, 467, 531, 563};
    static const int feCo[6] = {128, 128,  96,  64,  32,   2};
    ConvW cFe[6];
    for (int i = 0; i < 6; ++i) cFe[i] = mkConv(ws, stream, fe_w[i], fe_b[i], feCi[i], feCo[i], 3);

    static const int ctxCi[7]  = {565, 128, 128, 128, 96, 64, 32};
    static const int ctxCo[7]  = {128, 128, 128,  96, 64, 32,  2};
    static const int ctxDil[7] = {1, 2, 4, 8, 16, 1, 1};
    ConvW cCtx[7];
    for (int i = 0; i < 7; ++i) cCtx[i] = mkConv(ws, stream, ctx_w[i], ctx_b[i], ctxCi[i], ctxCo[i], 3);

    static const int segCi[6] = {192, 160, 128, 96, 64, 48};
    ConvW cSegE[6], cSegP[6];
    for (int i = 0; i < 6; ++i) {
        cSegE[i] = mkConv(ws, stream, seg[i].exp_w,  nullptr, segCi[i], segCi[i], 1);
        cSegP[i] = mkConv(ws, stream, seg[i].proj_w, nullptr, segCi[i], 32, 1);
    }
    ConvW cPw1 = mkConv(ws, stream, d1.pw_w, nullptr, 32, 32, 1);
    ConvW cPw2 = mkConv(ws, stream, d2.pw_w, nullptr, 32, 32, 1);
    ConvW cOut = mkConv(ws, stream, cls_out_w, cls_out_b, 32, 19, 1);

    // -------- feature pyramids (both images), fine->coarse then reversed
    float* p1s[6]; float* p2s[6];
    int lH[6], lW[6];
    {
        const float* x1 = im1; const float* x2 = im2;
        int Hc = H0, Wc = W0, curC = 3;
        for (int i = 0; i < 6; ++i) {
            int H2 = outDim(Hc, 3, 2, 1), W2 = outDim(Wc, 3, 2, 1);
            int co = chs[i + 1];
            float* t1 = ws.f((size_t)B * co * H2 * W2);
            float* o1 = ws.f((size_t)B * co * H2 * W2);
            runConv(stream, cFpe1[i], x1, curC, 0, Hc, Wc, t1, co, 0, B, 2, 1, nullptr, nullptr, ACT_LRELU);
            runConv(stream, cFpe2[i], t1, co, 0, H2, W2, o1, co, 0, B, 1, 1, nullptr, nullptr, ACT_LRELU);
            float* t2 = ws.f((size_t)B * co * H2 * W2);
            float* o2 = ws.f((size_t)B * co * H2 * W2);
            runConv(stream, cFpe1[i], x2, curC, 0, Hc, Wc, t2, co, 0, B, 2, 1, nullptr, nullptr, ACT_LRELU);
            runConv(stream, cFpe2[i], t2, co, 0, H2, W2, o2, co, 0, B, 1, 1, nullptr, nullptr, ACT_LRELU);
            p1s[i] = o1; p2s[i] = o2;
            x1 = o1; x2 = o2; curC = co; Hc = H2; Wc = W2;
            lH[i] = H2; lW[i] = W2;
        }
    }
    // reversed views (coarsest first)
    const float* pyr1[6]; const float* pyr2[6];
    int rH[6], rW[6], rC[6];
    for (int l = 0; l < 6; ++l) {
        pyr1[l] = p1s[5 - l]; pyr2[l] = p2s[5 - l];
        rH[l] = lH[5 - l]; rW[l] = lW[5 - l]; rC[l] = chs[6 - l];
    }

    float* outF = (float*)d_out;
    const size_t segN = (size_t)B * 19 * H0 * W0;

    // -------- segmentation head
    {
        const float* x = pyr1[0];
        int xCt = 192, Hc = rH[0], Wc = rW[0];
        float* xbuf = nullptr;
        for (int lvl = 0; lvl < 6; ++lvl) {
            int ci = segCi[lvl];
            size_t hw = (size_t)Hc * Wc;
            float* t1 = ws.f((size_t)B * ci * hw);
            runConv(stream, cSegE[lvl], x, xCt, 0, Hc, Wc, t1, ci, 0, B, 1, 1,
                    seg[lvl].bne_g, seg[lvl].bne_b, ACT_RELU);
            float* t2 = ws.f((size_t)B * ci * hw);
            k_dwconv3x3<<<g1((size_t)B * ci * hw), 256, 0, stream>>>(
                t1, ci, 0, seg[lvl].dw_w, seg[lvl].bnd_g, seg[lvl].bnd_b,
                t2, ci, 0, B, ci, Hc, Wc, ACT_RELU);
            float* t3 = ws.f((size_t)B * 32 * hw);
            runConv(stream, cSegP[lvl], t2, ci, 0, Hc, Wc, t3, 32, 0, B, 1, 1,
                    seg[lvl].bnp_g, seg[lvl].bnp_b, ACT_NONE);
            int Hn = Hc * 2, Wn = Wc * 2;
            int cp = (lvl < 5) ? rC[lvl + 1] : 0;
            int cn = 32 + cp;
            xbuf = ws.f((size_t)B * cn * Hn * Wn);
            k_nearest2x<<<g1((size_t)B * 32 * Hn * Wn), 256, 0, stream>>>(
                t3, 32, 0, xbuf, cn, 0, B, 32, Hc, Wc);
            if (lvl < 5)
                k_copy_ch<<<g1((size_t)B * cp * Hn * Wn), 256, 0, stream>>>(
                    pyr1[lvl + 1], cp, 0, xbuf, cn, 32, B, cp, Hn, Wn);
            x = xbuf; xCt = cn; Hc = Hn; Wc = Wn;
        }
        // classifier at 192x640 on 32 channels
        size_t hw = (size_t)Hc * Wc;
        float* u1 = ws.f((size_t)B * 32 * hw);
        float* u2 = ws.f((size_t)B * 32 * hw);
        k_dwconv3x3<<<g1((size_t)B * 32 * hw), 256, 0, stream>>>(
            x, 32, 0, d1.dw_w, d1.bn1_g, d1.bn1_b, u1, 32, 0, B, 32, Hc, Wc, ACT_RELU);
        runConv(stream, cPw1, u1, 32, 0, Hc, Wc, u2, 32, 0, B, 1, 1, d1.bn2_g, d1.bn2_b, ACT_RELU);
        k_dwconv3x3<<<g1((size_t)B * 32 * hw), 256, 0, stream>>>(
            u2, 32, 0, d2.dw_w, d2.bn1_g, d2.bn1_b, u1, 32, 0, B, 32, Hc, Wc, ACT_RELU);
        runConv(stream, cPw2, u1, 32, 0, Hc, Wc, u2, 32, 0, B, 1, 1, d2.bn2_g, d2.bn2_b, ACT_RELU);
        runConv(stream, cOut, u2, 32, 0, Hc, Wc, outF, 19, 0, B, 1, 1, nullptr, nullptr, ACT_NONE);
    }

    // -------- shared flow-stage buffers (sized for level 4: 48x160)
    const size_t maxHW = (size_t)rH[4] * rW[4];     // 7680
    float* E     = ws.f((size_t)B * 565 * maxHW);   // fused concat: feat[0..563) | flow[563..565)
    float* ctxA  = ws.f((size_t)B * 128 * maxHW);
    float* ctxB  = ws.f((size_t)B * 128 * maxHW);
    float* im2w  = ws.f((size_t)B * 32  * maxHW);   // max C*H*W over levels = 245760/batch
    float* fres  = ws.f((size_t)B * 2   * maxHW);
    float* flowL[5];
    for (int l = 0; l < 5; ++l) flowL[l] = ws.f((size_t)B * 2 * rH[l] * rW[l]);

    size_t flowElems = 0;
    for (int l = 0; l < 5; ++l) flowElems += (size_t)B * 2 * (4 * rH[l]) * (4 * rW[l]);

    auto runAux = [&](const float* const* pa, const float* const* pb, size_t outBase) {
        for (int l = 0; l < 5; ++l) {
            const int Hc = rH[l], Wc = rW[l], Cl = rC[l];
            const size_t hw = (size_t)Hc * Wc;
            float* flow = flowL[l];
            const float* imA = pa[l];
            const float* imB = pb[l];
            const float* im2wp;
            if (l == 0) {
                k_fill<<<g1((size_t)B * 2 * hw), 256, 0, stream>>>(flow, 0.f, (int)(B * 2 * hw));
                im2wp = imB;
            } else {
                k_resize_ac<<<g1((size_t)B * 2 * hw), 256, 0, stream>>>(
                    flowL[l - 1], flow, B, 2, rH[l - 1], rW[l - 1], Hc, Wc, 2.0f);
                k_warp<<<g1((size_t)B * hw), 256, 0, stream>>>(imB, flow, im2w, B, Cl, Hc, Wc);
                im2wp = im2w;
            }
            // corr (81ch) -> E@448 ; 1x1 conv (32ch) -> E@529 ; flow -> E@561
            k_corr81<<<g1((size_t)B * hw), 256, 0, stream>>>(imA, im2wp, E, 565, 448, B, Cl, Hc, Wc);
            runConv(stream, cC11[l], imA, Cl, 0, Hc, Wc, E, 565, 529, B, 1, 1, nullptr, nullptr, ACT_LRELU);
            k_copy_ch<<<g1((size_t)B * 2 * hw), 256, 0, stream>>>(flow, 2, 0, E, 565, 561, B, 2, Hc, Wc);
            // FlowEstimatorDense: prepend-concat = descending channel offsets in E
            runConv(stream, cFe[0], E, 565, 448, Hc, Wc, E, 565, 320, B, 1, 1, nullptr, nullptr, ACT_LRELU);
            runConv(stream, cFe[1], E, 565, 320, Hc, Wc, E, 565, 192, B, 1, 1, nullptr, nullptr, ACT_LRELU);
            runConv(stream, cFe[2], E, 565, 192, Hc, Wc, E, 565,  96, B, 1, 1, nullptr, nullptr, ACT_LRELU);
            runConv(stream, cFe[3], E, 565,  96, Hc, Wc, E, 565,  32, B, 1, 1, nullptr, nullptr, ACT_LRELU);
            runConv(stream, cFe[4], E, 565,  32, Hc, Wc, E, 565,   0, B, 1, 1, nullptr, nullptr, ACT_LRELU);
            runConv(stream, cFe[5], E, 565,   0, Hc, Wc, fres, 2, 0, B, 1, 1, nullptr, nullptr, ACT_NONE);
            k_add<<<g1((size_t)B * 2 * hw), 256, 0, stream>>>(flow, fres, (int)(B * 2 * hw));
            // context net on cat([feat(0..563), flow]) = E[0..565)
            k_copy_ch<<<g1((size_t)B * 2 * hw), 256, 0, stream>>>(flow, 2, 0, E, 565, 563, B, 2, Hc, Wc);
            runConv(stream, cCtx[0], E,    565, 0, Hc, Wc, ctxA, 128, 0, B, 1, ctxDil[0], nullptr, nullptr, ACT_LRELU);
            runConv(stream, cCtx[1], ctxA, 128, 0, Hc, Wc, ctxB, 128, 0, B, 1, ctxDil[1], nullptr, nullptr, ACT_LRELU);
            runConv(stream, cCtx[2], ctxB, 128, 0, Hc, Wc, ctxA, 128, 0, B, 1, ctxDil[2], nullptr, nullptr, ACT_LRELU);
            runConv(stream, cCtx[3], ctxA, 128, 0, Hc, Wc, ctxB,  96, 0, B, 1, ctxDil[3], nullptr, nullptr, ACT_LRELU);
            runConv(stream, cCtx[4], ctxB,  96, 0, Hc, Wc, ctxA,  64, 0, B, 1, ctxDil[4], nullptr, nullptr, ACT_LRELU);
            runConv(stream, cCtx[5], ctxA,  64, 0, Hc, Wc, ctxB,  32, 0, B, 1, ctxDil[5], nullptr, nullptr, ACT_LRELU);
            runConv(stream, cCtx[6], ctxB,  32, 0, Hc, Wc, fres,   2, 0, B, 1, ctxDil[6], nullptr, nullptr, ACT_NONE);
            k_add<<<g1((size_t)B * 2 * hw), 256, 0, stream>>>(flow, fres, (int)(B * 2 * hw));
        }
        // outputs: flows[::-1], each resized x4 (values scaled x4), concatenated
        size_t off = outBase;
        for (int k = 0; k < 5; ++k) {
            int l = 4 - k;
            int Ho = 4 * rH[l], Wo = 4 * rW[l];
            size_t n = (size_t)B * 2 * Ho * Wo;
            k_resize_ac<<<g1(n), 256, 0, stream>>>(
                flowL[l], outF + off, B, 2, rH[l], rW[l], Ho, Wo, 4.0f);
            off += n;
        }
    };

    runAux(pyr1, pyr2, segN);               // forward flows
    runAux(pyr2, pyr1, segN + flowElems);   // backward flows
}
